// RepFlowLayer_84327387889858
// MI455X (gfx1250) — compile-verified, matrix operand order checked by reading, not run on hardware
//
#include <hip/hip_runtime.h>
#include <hip/hip_bf16.h>

typedef __attribute__((ext_vector_type(16))) __bf16       v16bf;
typedef __attribute__((ext_vector_type(8)))  float        v8f;
typedef __attribute__((ext_vector_type(4)))  unsigned int u32x4;

#define NLOC 512
#define NALL 640
#define NNEI 96
#define ASEL 20
#define NDIM 128
#define EDIM 64
#define ADIM 64
#define KEDGE 320
#define KANG  320

// weight-fragment table (each fragment = 32 lanes * 16 bf16 = 512 ushort)
#define FRAG_EDGE 0      // 10 ksteps x 12 ntiles  (N=192: [w_edge1|w_lin2])
#define FRAG_PROJ 120    //  2 ksteps x  8 ntiles  (N=128: w_proj)
#define FRAG_ANG  136    // 10 ksteps x  8 ntiles  (N=128: [w_ang|w_ga1])
#define FRAG_GA2  216    //  2 ksteps x  4 ntiles  (N= 64: w_ga2)
#define NFRAG 224

__device__ __forceinline__ unsigned short f2bf(float f) {
    union { float f; unsigned int u; } x; x.f = f;
    unsigned int r = x.u + 0x7FFFu + ((x.u >> 16) & 1u);
    return (unsigned short)(r >> 16);
}
__device__ __forceinline__ float bf2f(unsigned short h) {
    union { unsigned int u; float f; } x; x.u = ((unsigned int)h) << 16;
    return x.f;
}
__device__ __forceinline__ float silu_f(float x) { return x / (1.0f + __expf(-x)); }

__device__ __forceinline__ v16bf frag_a(const unsigned short* row, int k0) {
    // A fragment: this lane needs K = {k0..k0+7, k0+16..k0+23} (row-major LDS, 16B aligned)
    union U { u32x4 u[2]; v16bf v; } x;
    x.u[0] = *(const u32x4*)(row + k0);
    x.u[1] = *(const u32x4*)(row + k0 + 16);
    return x.v;
}
__device__ __forceinline__ v16bf frag_b(const unsigned short* wb, int fragIdx, int lane) {
    union U { u32x4 u[2]; v16bf v; } x;
    const u32x4* p = (const u32x4*)(wb + (size_t)fragIdx * 512 + lane * 16);
    x.u[0] = p[0]; x.u[1] = p[1];
    return x.v;
}
__device__ __forceinline__ v8f wmma_bf16(v16bf a, v16bf b, v8f c) {
    return __builtin_amdgcn_wmma_f32_16x16x32_bf16(false, a, false, b, (short)0, c, false, false);
}

// ---------------------------------------------------------------------------
// prep: convert weights fp32 -> bf16 WMMA-B fragments in ws
// grid 224 x 512 ; block b = fragment, thread = (lane, p)
// ---------------------------------------------------------------------------
__global__ __launch_bounds__(512) void k_prep(const float* __restrict__ w_edge1,
                                              const float* __restrict__ w_lin2,
                                              const float* __restrict__ w_proj,
                                              const float* __restrict__ w_ang,
                                              const float* __restrict__ w_ga1,
                                              const float* __restrict__ w_ga2,
                                              unsigned short* __restrict__ wb) {
    int frag = blockIdx.x;
    int t = threadIdx.x >> 4;          // lane 0..31
    int p = threadIdx.x & 15;          // element 0..15 within lane
    int k_in = (p < 8 ? p : p + 8) + (t < 16 ? 0 : 8);
    int nloc = t & 15;
    int seg, ks, nt;
    if (frag < FRAG_PROJ)      { seg = 0; ks = frag / 12;            nt = frag % 12; }
    else if (frag < FRAG_ANG)  { seg = 1; ks = (frag - FRAG_PROJ)/8; nt = (frag - FRAG_PROJ)%8; }
    else if (frag < FRAG_GA2)  { seg = 2; ks = (frag - FRAG_ANG)/8;  nt = (frag - FRAG_ANG)%8; }
    else                       { seg = 3; ks = (frag - FRAG_GA2)/4;  nt = (frag - FRAG_GA2)%4; }
    int k = ks * 32 + k_in;
    int n = nt * 16 + nloc;
    float v;
    if (seg == 0)      v = (n < 128) ? w_edge1[k * 128 + n] : w_lin2[k * 64 + (n - 128)];
    else if (seg == 1) v = w_proj[k * 128 + n];
    else if (seg == 2) v = (n < 64) ? w_ang[k * 64 + n] : w_ga1[k * 64 + (n - 64)];
    else               v = w_ga2[k * 64 + n];
    wb[(size_t)frag * 512 + t * 16 + p] = f2bf(v);
}

// ---------------------------------------------------------------------------
// edge kernel: one WG per atom l.  M=96 (6 mtiles), fused N=192 GEMM + proj.
// ---------------------------------------------------------------------------
__global__ __launch_bounds__(256) void k_edge(const float* __restrict__ g1_ext,
                                              const float* __restrict__ g2,
                                              const float* __restrict__ sw,
                                              const unsigned char* __restrict__ nmask,
                                              const int* __restrict__ nlist,
                                              const unsigned short* __restrict__ wb,
                                              const float* __restrict__ b_edge1,
                                              const float* __restrict__ b_lin2,
                                              float* __restrict__ out_g2,
                                              float* __restrict__ edge_sum,
                                              float* __restrict__ proj_sum) {
    __shared__ __align__(16) unsigned short lA[NNEI][KEDGE];   // edge_info rows, bf16
    __shared__ float lmsw[NNEI];
    __shared__ float lred[2 * NDIM];
    const float RSQRT3 = 0.57735026919f;
    int l = blockIdx.x;
    int tid = threadIdx.x;

    for (int n = tid; n < NNEI; n += 256)
        lmsw[n] = (nmask[l * NNEI + n] ? 1.0f : 0.0f) * sw[l * NNEI + n];
    for (int i = tid; i < 2 * NDIM; i += 256) lred[i] = 0.0f;

    // stage edge_info = [g1(128) | gg1(128) | g2(64)] as bf16
    for (int idx = tid; idx < NNEI * KEDGE; idx += 256) {
        int n = idx / KEDGE, k = idx % KEDGE;
        float v;
        if (k < NDIM)            v = g1_ext[(size_t)l * NDIM + k];
        else if (k < 2 * NDIM)   v = g1_ext[(size_t)nlist[l * NNEI + n] * NDIM + (k - NDIM)];
        else                     v = g2[((size_t)l * NNEI + n) * EDIM + (k - 2 * NDIM)];
        lA[n][k] = f2bf(v);
    }
    __syncthreads();

    int wave = tid >> 5, lane = tid & 31;
    int half = lane >> 4, lmod = lane & 15;

    if (wave < 6) {
        // fused [w_edge1 | w_lin2] GEMM: this wave owns ntiles {2w, 2w+1}
        for (int mt = 0; mt < 6; ++mt) {
            v8f acc[2];
            acc[0] = v8f{0.f,0.f,0.f,0.f,0.f,0.f,0.f,0.f};
            acc[1] = acc[0];
            for (int ks = 0; ks < 10; ++ks) {
                int row = mt * 16 + lmod;
                v16bf av = frag_a(&lA[row][0], ks * 32 + half * 8);
                for (int t = 0; t < 2; ++t) {
                    int nt = 2 * wave + t;
                    v16bf bv = frag_b(wb, FRAG_EDGE + ks * 12 + nt, lane);
                    acc[t] = wmma_bf16(av, bv, acc[t]);
                }
            }
            for (int t = 0; t < 2; ++t) {
                int col = (2 * wave + t) * 16 + lmod;
                if (col < NDIM) {
                    float colsum = 0.0f;
                    for (int v = 0; v < 8; ++v) {
                        int row = mt * 16 + half * 8 + v;
                        colsum += silu_f(acc[t][v] + b_edge1[col]) * lmsw[row];
                    }
                    atomicAdd(&lred[col], colsum);
                } else {
                    int c2 = col - NDIM;
                    for (int v = 0; v < 8; ++v) {
                        int row = mt * 16 + half * 8 + v;
                        size_t o = ((size_t)l * NNEI + row) * EDIM + c2;
                        out_g2[o] = (g2[o] + silu_f(acc[t][v] + b_lin2[c2])) * RSQRT3;
                    }
                }
            }
        }
    } else {
        // proj GEMM: A = g2 slice of LDS (K at offset 256), N=128, waves 6,7 own 4 ntiles each
        int wp = wave - 6;
        for (int mt = 0; mt < 6; ++mt) {
            v8f acc[4];
            acc[0] = v8f{0.f,0.f,0.f,0.f,0.f,0.f,0.f,0.f};
            acc[1] = acc[0]; acc[2] = acc[0]; acc[3] = acc[0];
            for (int ks = 0; ks < 2; ++ks) {
                int row = mt * 16 + lmod;
                v16bf av = frag_a(&lA[row][0], 256 + ks * 32 + half * 8);
                for (int t = 0; t < 4; ++t) {
                    int nt = wp * 4 + t;
                    v16bf bv = frag_b(wb, FRAG_PROJ + ks * 8 + nt, lane);
                    acc[t] = wmma_bf16(av, bv, acc[t]);
                }
            }
            for (int t = 0; t < 4; ++t) {
                int col = (wp * 4 + t) * 16 + lmod;
                float s = 0.0f;
                for (int v = 0; v < 8; ++v) {
                    int row = mt * 16 + half * 8 + v;
                    s += acc[t][v] * bf2f(lA[row][NDIM + col]) * lmsw[row];
                }
                atomicAdd(&lred[NDIM + col], s);
            }
        }
    }
    __syncthreads();
    const float INV96 = 1.0f / 96.0f;
    for (int c = tid; c < NDIM; c += 256) {
        edge_sum[(size_t)l * NDIM + c] = lred[c] * INV96;
        proj_sum[(size_t)l * NDIM + c] = lred[NDIM + c] * INV96;
    }
}

// ---------------------------------------------------------------------------
// angle kernel: one WG per (l, i). M=20(pad 32), K=320, N=128 = [w_ang|w_ga1]
// ---------------------------------------------------------------------------
__global__ __launch_bounds__(128) void k_angle(const float* __restrict__ g1_ext,
                                               const float* __restrict__ g2,
                                               const float* __restrict__ angle_embed,
                                               const float* __restrict__ a_sw,
                                               const unsigned char* __restrict__ amask,
                                               const unsigned short* __restrict__ wb,
                                               const float* __restrict__ b_ang,
                                               const float* __restrict__ b_ga1,
                                               float* __restrict__ out_anew,
                                               float* __restrict__ g2_ang) {
    __shared__ __align__(16) unsigned short lA[32][KANG];      // ang rows, bf16
    __shared__ float lscale[32];
    __shared__ float lg2ang[EDIM];
    int blk = blockIdx.x;
    int l = blk / ASEL, i = blk % ASEL;
    int tid = threadIdx.x;
    float aswi = a_sw[l * ASEL + i];
    float mi = amask[l * ASEL + i] ? 1.0f : 0.0f;

    // ang row j = [angle_embed(64) | g1(128) | g2a_i(64) | g2a_j(64)], rows 20..31 = 0
    for (int idx = tid; idx < 32 * KANG; idx += 128) {
        int j = idx / KANG, k = idx % KANG;
        float v = 0.0f;
        if (j < ASEL) {
            if (k < ADIM)            v = angle_embed[(((size_t)l * ASEL + i) * ASEL + j) * ADIM + k];
            else if (k < ADIM + NDIM) v = g1_ext[(size_t)l * NDIM + (k - ADIM)];
            else if (k < ADIM + NDIM + EDIM)
                v = mi * g2[((size_t)l * NNEI + i) * EDIM + (k - ADIM - NDIM)];
            else {
                float mj = amask[l * ASEL + j] ? 1.0f : 0.0f;
                v = mj * g2[((size_t)l * NNEI + j) * EDIM + (k - ADIM - NDIM - EDIM)];
            }
        }
        lA[j][k] = f2bf(v);
    }
    for (int j = tid; j < 32; j += 128)
        lscale[j] = (j < ASEL) ? aswi * a_sw[l * ASEL + j] : 0.0f;
    if (tid < EDIM) lg2ang[tid] = 0.0f;
    __syncthreads();

    int wave = tid >> 5, lane = tid & 31;
    int half = lane >> 4, lmod = lane & 15;

    v8f acc[2][2];
    acc[0][0] = v8f{0.f,0.f,0.f,0.f,0.f,0.f,0.f,0.f};
    acc[0][1] = acc[0][0]; acc[1][0] = acc[0][0]; acc[1][1] = acc[0][0];
    for (int ks = 0; ks < 10; ++ks) {
        v16bf av[2];
        for (int mt = 0; mt < 2; ++mt)
            av[mt] = frag_a(&lA[mt * 16 + lmod][0], ks * 32 + half * 8);
        for (int t = 0; t < 2; ++t) {
            int nt = 2 * wave + t;
            v16bf bv = frag_b(wb, FRAG_ANG + ks * 8 + nt, lane);
            for (int mt = 0; mt < 2; ++mt)
                acc[mt][t] = wmma_bf16(av[mt], bv, acc[mt][t]);
        }
    }

    const float RSQRT2 = 0.70710678119f;
    for (int mt = 0; mt < 2; ++mt) {
        for (int t = 0; t < 2; ++t) {
            int col = (2 * wave + t) * 16 + lmod;
            if (col < ADIM) {
                for (int v = 0; v < 8; ++v) {
                    int j = mt * 16 + half * 8 + v;
                    if (j < ASEL) {
                        size_t o = (((size_t)l * ASEL + i) * ASEL + j) * ADIM + col;
                        out_anew[o] = (angle_embed[o] + silu_f(acc[mt][t][v] + b_ang[col])) * RSQRT2;
                    }
                }
            } else {
                int c = col - ADIM;
                float s = 0.0f;
                for (int v = 0; v < 8; ++v) {
                    int j = mt * 16 + half * 8 + v;
                    if (j < ASEL) s += silu_f(lscale[j] * acc[mt][t][v] + b_ga1[c]);
                }
                atomicAdd(&lg2ang[c], s);
            }
        }
    }
    __syncthreads();
    if (tid < EDIM)
        g2_ang[((size_t)l * ASEL + i) * EDIM + tid] = lg2ang[tid] * (1.0f / (float)ASEL);
}

// ---------------------------------------------------------------------------
// ga2 kernel: g2_ang_full @ w_ga2; rows >= ASEL contribute silu(b_ga2) const.
// ---------------------------------------------------------------------------
__global__ __launch_bounds__(128) void k_ga2(const float* __restrict__ g2_ang,
                                             const unsigned short* __restrict__ wb,
                                             const float* __restrict__ b_ga2,
                                             float* __restrict__ out_g2) {
    __shared__ __align__(16) unsigned short lA[32][EDIM];
    int l = blockIdx.x, tid = threadIdx.x;
    const float RSQRT3 = 0.57735026919f;

    for (int idx = tid; idx < 32 * EDIM; idx += 128) {
        int j = idx / EDIM, k = idx % EDIM;
        float v = (j < ASEL) ? g2_ang[((size_t)l * ASEL + j) * EDIM + k] : 0.0f;
        lA[j][k] = f2bf(v);
    }
    __syncthreads();

    int wave = tid >> 5, lane = tid & 31;
    int half = lane >> 4, lmod = lane & 15;

    v8f acc[2];
    acc[0] = v8f{0.f,0.f,0.f,0.f,0.f,0.f,0.f,0.f};
    acc[1] = acc[0];
    for (int ks = 0; ks < 2; ++ks) {
        for (int mt = 0; mt < 2; ++mt) {
            v16bf av = frag_a(&lA[mt * 16 + lmod][0], ks * 32 + half * 8);
            v16bf bv = frag_b(wb, FRAG_GA2 + ks * 4 + wave, lane);
            acc[mt] = wmma_bf16(av, bv, acc[mt]);
        }
    }
    int col = wave * 16 + lmod;
    for (int mt = 0; mt < 2; ++mt) {
        for (int v = 0; v < 8; ++v) {
            int j = mt * 16 + half * 8 + v;
            if (j < ASEL) {
                size_t o = ((size_t)l * NNEI + j) * EDIM + col;
                out_g2[o] += silu_f(acc[mt][v] + b_ga2[col]) * RSQRT3;
            }
        }
    }
    // tail rows: g2_ang_full row == 0 -> silu(bias) constant
    for (int idx = tid; idx < (NNEI - ASEL) * EDIM; idx += 128) {
        int j = ASEL + idx / EDIM, c = idx % EDIM;
        out_g2[((size_t)l * NNEI + j) * EDIM + c] += silu_f(b_ga2[c]) * RSQRT3;
    }
}

// ---------------------------------------------------------------------------
// g1 final: sym op (VALU), lin1, self, combine; h2 passthrough.
// ---------------------------------------------------------------------------
__global__ __launch_bounds__(128) void k_g1final(const float* __restrict__ g1_ext,
                                                 const float* __restrict__ g2,
                                                 const float* __restrict__ h2,
                                                 const float* __restrict__ sw,
                                                 const unsigned char* __restrict__ nmask,
                                                 const int* __restrict__ nlist,
                                                 const float* __restrict__ w_self,
                                                 const float* __restrict__ b_self,
                                                 const float* __restrict__ w_lin1,
                                                 const float* __restrict__ b_lin1,
                                                 const float* __restrict__ edge_sum,
                                                 const float* __restrict__ proj_sum,
                                                 float* __restrict__ out_g1,
                                                 float* __restrict__ out_h2) {
    __shared__ float hg[3][192];     // [t][0:64]=hg(g2), [t][64:192]=hg(gg1)
    __shared__ float sym[768];
    __shared__ float lmsw[NNEI];
    int l = blockIdx.x, tid = threadIdx.x;
    for (int n = tid; n < NNEI; n += 128)
        lmsw[n] = (nmask[l * NNEI + n] ? 1.0f : 0.0f) * sw[l * NNEI + n];
    __syncthreads();

    const float INVS = 0.10206207262f;   // 1/sqrt(96)
    for (int idx = tid; idx < 3 * 192; idx += 128) {
        int t = idx / 192, d = idx % 192;
        float s = 0.0f;
        if (d < EDIM) {
            for (int n = 0; n < NNEI; ++n)
                s += h2[((size_t)l * NNEI + n) * 3 + t] * g2[((size_t)l * NNEI + n) * EDIM + d] * lmsw[n];
        } else {
            int dd = d - EDIM;
            for (int n = 0; n < NNEI; ++n)
                s += h2[((size_t)l * NNEI + n) * 3 + t] *
                     g1_ext[(size_t)nlist[l * NNEI + n] * NDIM + dd] * lmsw[n];
        }
        hg[t][d] = s * INVS;
    }
    __syncthreads();
    for (int idx = tid; idx < 768; idx += 128) {
        float s;
        if (idx < 256) {
            int d = idx >> 2, a = idx & 3;
            s = hg[0][d]*hg[0][a] + hg[1][d]*hg[1][a] + hg[2][d]*hg[2][a];
        } else {
            int r = idx - 256, d = r >> 2, a = r & 3;
            s = hg[0][64+d]*hg[0][64+a] + hg[1][64+d]*hg[1][64+a] + hg[2][64+d]*hg[2][64+a];
        }
        sym[idx] = s;
    }
    __syncthreads();
    const float RSQRT5 = 0.44721359550f;
    for (int c = tid; c < NDIM; c += 128) {
        float a_self = b_self[c];
        for (int k = 0; k < NDIM; ++k) a_self += g1_ext[(size_t)l * NDIM + k] * w_self[k * NDIM + c];
        float a_lin1 = b_lin1[c];
        for (int k = 0; k < 768; ++k) a_lin1 += sym[k] * w_lin1[k * NDIM + c];
        float g1v = g1_ext[(size_t)l * NDIM + c];
        out_g1[(size_t)l * NDIM + c] =
            (g1v + silu_f(a_self) + proj_sum[(size_t)l * NDIM + c] + silu_f(a_lin1) +
             edge_sum[(size_t)l * NDIM + c]) * RSQRT5;
    }
    for (int idx = tid; idx < NNEI * 3; idx += 128)
        out_h2[(size_t)l * NNEI * 3 + idx] = h2[(size_t)l * NNEI * 3 + idx];
}

// ---------------------------------------------------------------------------
extern "C" void kernel_launch(void* const* d_in, const int* in_sizes, int n_in,
                              void* d_out, int out_size, void* d_ws, size_t ws_size,
                              hipStream_t stream) {
    const float* g1_ext      = (const float*)d_in[0];
    const float* g2          = (const float*)d_in[1];
    const float* h2          = (const float*)d_in[2];
    const float* angle_embed = (const float*)d_in[3];
    const float* sw          = (const float*)d_in[4];
    const float* a_sw        = (const float*)d_in[5];
    const float* w_self      = (const float*)d_in[6];
    const float* b_self      = (const float*)d_in[7];
    const float* w_proj      = (const float*)d_in[8];
    const float* w_lin1      = (const float*)d_in[9];
    const float* b_lin1      = (const float*)d_in[10];
    const float* w_edge1     = (const float*)d_in[11];
    const float* b_edge1     = (const float*)d_in[12];
    const float* w_lin2      = (const float*)d_in[13];
    const float* b_lin2      = (const float*)d_in[14];
    const float* w_ang       = (const float*)d_in[15];
    const float* b_ang       = (const float*)d_in[16];
    const float* w_ga1       = (const float*)d_in[17];
    const float* b_ga1       = (const float*)d_in[18];
    const float* w_ga2       = (const float*)d_in[19];
    const float* b_ga2       = (const float*)d_in[20];
    const int*   nlist       = (const int*)d_in[21];
    const unsigned char* nmask = (const unsigned char*)d_in[22];
    const unsigned char* amask = (const unsigned char*)d_in[23];

    float* out      = (float*)d_out;
    float* out_g1   = out;                                   // 512*128
    float* out_g2   = out + (size_t)NLOC * NDIM;             // 512*96*64
    float* out_h2   = out_g2 + (size_t)NLOC * NNEI * EDIM;   // 512*96*3
    float* out_anew = out_h2 + (size_t)NLOC * NNEI * 3;      // 512*20*20*64

    unsigned short* wb = (unsigned short*)d_ws;              // NFRAG*512 ushort
    float* edge_sum = (float*)((char*)d_ws + (size_t)NFRAG * 512 * sizeof(unsigned short));
    float* proj_sum = edge_sum + (size_t)NLOC * NDIM;
    float* g2_ang   = proj_sum + (size_t)NLOC * NDIM;        // 512*20*64

    k_prep<<<NFRAG, 512, 0, stream>>>(w_edge1, w_lin2, w_proj, w_ang, w_ga1, w_ga2, wb);
    k_edge<<<NLOC, 256, 0, stream>>>(g1_ext, g2, sw, nmask, nlist, wb, b_edge1, b_lin2,
                                     out_g2, edge_sum, proj_sum);
    k_angle<<<NLOC * ASEL, 128, 0, stream>>>(g1_ext, g2, angle_embed, a_sw, amask, wb,
                                             b_ang, b_ga1, out_anew, g2_ang);
    k_ga2<<<NLOC, 128, 0, stream>>>(g2_ang, wb, b_ga2, out_g2);
    k_g1final<<<NLOC, 128, 0, stream>>>(g1_ext, g2, h2, sw, nmask, nlist, w_self, b_self,
                                        w_lin1, b_lin1, edge_sum, proj_sum, out_g1, out_h2);
}
